// TemporalGCN_53609781788683
// MI455X (gfx1250) — matrix operand order repaired and checked by reading, not compiled, not run on hardware
//
#include <hip/hip_runtime.h>
#include <hip/hip_bf16.h>
#include <stdint.h>

// Problem constants (from reference)
#define DCH    128   // in_channels
#define HCH    128   // hidden_dim
#define GB     128   // graphs
#define LHD    128   // lstm hidden
#define NGATE  512   // 4*LH
#define OUTD   16
#define NLAYER 3

typedef __attribute__((ext_vector_type(16))) __bf16          v16bf;
typedef __attribute__((ext_vector_type(16))) unsigned short  v16u;
typedef __attribute__((ext_vector_type(8)))  unsigned short  v8us;
typedef __attribute__((ext_vector_type(8)))  unsigned int    v8u;
typedef __attribute__((ext_vector_type(8)))  float           v8f;

// split fp32 -> bf16 hi + bf16 lo (a ~= hi + lo), truncation split
__device__ __forceinline__ void splitbf(float a, unsigned short& h, unsigned short& l) {
  unsigned int u = __builtin_bit_cast(unsigned int, a);
  unsigned short hs = (unsigned short)(u >> 16);
  float hf = __builtin_bit_cast(float, (unsigned int)hs << 16);
  unsigned int r = __builtin_bit_cast(unsigned int, a - hf);
  h = hs;
  l = (unsigned short)(r >> 16);
}

// ---------------- degree / normalization ----------------
__global__ void init_deg_kernel(float* deg, int n) {
  int i = blockIdx.x * blockDim.x + threadIdx.x;
  if (i < n) deg[i] = 1.0f;  // self-loop
}
__global__ void accum_deg_kernel(const int* __restrict__ dst, float* deg, int e) {
  int i = blockIdx.x * blockDim.x + threadIdx.x;
  if (i < e) atomicAdd(&deg[dst[i]], 1.0f);
}
__global__ void finish_dinv_kernel(float* deg, int n) {
  int i = blockIdx.x * blockDim.x + threadIdx.x;
  if (i < n) deg[i] = rsqrtf(fmaxf(deg[i], 1.0f));
}

// ---------------- weight splitting ----------------
// plain: hi/lo at same index (used for W_ih: B[k][n]=W_ih[n*128+k], col-major B == row-major W_ih)
__global__ void split_plain_kernel(const float* __restrict__ src,
                                   unsigned short* hi, unsigned short* lo, int n) {
  int i = blockIdx.x * blockDim.x + threadIdx.x;
  if (i < n) splitbf(src[i], hi[i], lo[i]);
}
// transpose per 128x128 layer: out[l][n*128+k] = src[l][k*128+n]  (col-major B for h@W)
__global__ void split_transpose_ws_kernel(const float* __restrict__ src,
                                          unsigned short* hi, unsigned short* lo) {
  int i = blockIdx.x * blockDim.x + threadIdx.x;
  if (i >= NLAYER * DCH * HCH) return;
  int l = i >> 14, r = i & 16383, n = r >> 7, k = r & 127;
  splitbf(src[(l << 14) + (k << 7) + n], hi[i], lo[i]);
}

// ---------------- WMMA GEMM: C[M x Ncols] = A[M x 128] @ B[128 x Ncols] ----------------
// BhiT/BloT stored column-major: [n*128 + k].
// Block = 8 waves: stages one 16x128 A-tile into LDS (pre-split to bf16 hi/lo, once
// per block instead of once per wave), then each wave owns one 16x16 output tile.
// bf16 hi/lo split: C ~= Ahi*Bhi + Ahi*Blo + Alo*Bhi  (3 WMMA per K=32 chunk, 12 per tile)
__global__ __launch_bounds__(256) void gemm_wmma_bf16x2_kernel(
    const float* __restrict__ A,
    const unsigned short* __restrict__ BhiT,
    const unsigned short* __restrict__ BloT,
    float* __restrict__ C, int M, int Ncols) {
  const int K = 128;
  __shared__ unsigned short sAhi[16][128];   // 4 KB
  __shared__ unsigned short sAlo[16][128];   // 4 KB
  int tilesN    = Ncols >> 4;
  int colGroups = (tilesN + 7) >> 3;         // 8 col-tiles per block
  int tM  = blockIdx.x / colGroups;
  int grp = blockIdx.x % colGroups;
  int m0  = tM << 4;
  int tid = threadIdx.x;

  // cooperative stage: 2048 elements, 8 per thread, split fp32 -> bf16 hi/lo
  {
    int base = tid * 8;
    int r = base >> 7, k = base & 127;
    const float* ap = A + (size_t)(m0 + r) * K + k;
    float4 f0 = *(const float4*)ap;
    float4 f1 = *(const float4*)(ap + 4);
    float av[8] = {f0.x, f0.y, f0.z, f0.w, f1.x, f1.y, f1.z, f1.w};
#pragma unroll
    for (int j = 0; j < 8; ++j) {
      unsigned short h, l;
      splitbf(av[j], h, l);
      sAhi[r][k + j] = h;
      sAlo[r][k + j] = l;
    }
  }
  __syncthreads();

  int wv = tid >> 5, lane = tid & 31;
  int tN = grp * 8 + wv;
  if (tN >= tilesN) return;   // wave-uniform: EXEC stays all-ones for WMMA
  int n0   = tN << 4;
  int half = lane >> 4;                  // 0: lanes 0-15, 1: lanes 16-31
  int col  = n0 + (lane & 15);
  int r    = lane & 15;
  const unsigned short* bhc = BhiT + (size_t)col * K;
  const unsigned short* blc = BloT + (size_t)col * K;

  v8f acc = {};
#pragma unroll
  for (int k0 = 0; k0 < K; k0 += 32) {
    // A fragment: lane half h holds k = k0+16h .. k0+16h+15 (matches B), two 16B runs from LDS per matrix
    int akb = k0 + half * 16;
    v8us ah0 = *(const v8us*)&sAhi[r][akb];
    v8us ah1 = *(const v8us*)&sAhi[r][akb + 8];
    v8us al0 = *(const v8us*)&sAlo[r][akb];
    v8us al1 = *(const v8us*)&sAlo[r][akb + 8];
    v16u ahU = __builtin_shufflevector(ah0, ah1, 0,1,2,3,4,5,6,7,8,9,10,11,12,13,14,15);
    v16u alU = __builtin_shufflevector(al0, al1, 0,1,2,3,4,5,6,7,8,9,10,11,12,13,14,15);
    // B fragment: lane holds one column; 16 contiguous bf16 (32B) per chunk from global
    int bkb = k0 + half * 16;
    v8u bh8 = *(const v8u*)(bhc + bkb);
    v8u bl8 = *(const v8u*)(blc + bkb);

    v16bf ah = __builtin_bit_cast(v16bf, ahU);
    v16bf al = __builtin_bit_cast(v16bf, alU);
    v16bf bh = __builtin_bit_cast(v16bf, bh8);
    v16bf bl = __builtin_bit_cast(v16bf, bl8);

    acc = __builtin_amdgcn_wmma_f32_16x16x32_bf16(false, ah, false, bh, (short)0, acc, false, false);
    acc = __builtin_amdgcn_wmma_f32_16x16x32_bf16(false, ah, false, bl, (short)0, acc, false, false);
    acc = __builtin_amdgcn_wmma_f32_16x16x32_bf16(false, al, false, bh, (short)0, acc, false, false);
  }
  // C/D layout: VGPR v -> row m0+v (lanes 0-15) or m0+8+v (lanes 16-31), col = n0+(lane&15)
  int rbase = m0 + half * 8;
#pragma unroll
  for (int v = 0; v < 8; ++v)
    C[(size_t)(rbase + v) * Ncols + col] = acc[v];
}

// ---------------- aggregation ----------------
__global__ void bias_init_kernel(float* agg, const float* __restrict__ bias, int n) {
  int i = blockIdx.x * blockDim.x + threadIdx.x;
  if (i < n) agg[i] = bias[i & (HCH - 1)];
}

// one wave (32 lanes) per edge (incl. self-loops); 4 floats per lane
__global__ void scatter_edges_kernel(const int* __restrict__ src, const int* __restrict__ dst,
                                     const float* __restrict__ dinv,
                                     const float* __restrict__ hW, float* agg,
                                     int e_cnt, int n_cnt) {
  unsigned int tid = blockIdx.x * blockDim.x + threadIdx.x;
  int e = (int)(tid >> 5);
  int lane = (int)(tid & 31);
  if (e >= e_cnt + n_cnt) return;
  int s, d;
  if (e < e_cnt) { s = src[e]; d = dst[e]; } else { s = e - e_cnt; d = s; }
  float norm = dinv[s] * dinv[d];
  const float* hs = hW + (size_t)s * HCH;
  __builtin_prefetch(hs, 0, 0);                       // global_prefetch_b8
  float4 v = *(const float4*)(hs + lane * 4);
  float* ap = agg + (size_t)d * HCH + lane * 4;
  atomicAdd(ap + 0, v.x * norm);
  atomicAdd(ap + 1, v.y * norm);
  atomicAdd(ap + 2, v.z * norm);
  atomicAdd(ap + 3, v.w * norm);
}

__global__ void bn_relu_kernel(const float* __restrict__ agg,
                               const float* __restrict__ gamma, const float* __restrict__ beta,
                               const float* __restrict__ rm, const float* __restrict__ rv,
                               float* out, int n) {
  int i = blockIdx.x * blockDim.x + threadIdx.x;
  if (i >= n) return;
  int j = i & (HCH - 1);
  float sc = gamma[j] * rsqrtf(rv[j] + 1e-5f);
  float y = (agg[i] - rm[j]) * sc + beta[j];
  out[i] = fmaxf(y, 0.0f);
}

// ---------------- pooling ----------------
__global__ void pool_zero_kernel(float* sums, float* cnt) {
  int i = blockIdx.x * blockDim.x + threadIdx.x;
  if (i < GB * HCH) sums[i] = 0.0f;
  else if (i < GB * HCH + GB) cnt[i - GB * HCH] = 0.0f;
}
__global__ void pool_accum_kernel(const float* __restrict__ h, const int* __restrict__ batch,
                                  float* sums, float* cnt, int n) {
  int i = blockIdx.x * blockDim.x + threadIdx.x;
  if (i >= n) return;
  int node = i >> 7, j = i & 127;
  int g = batch[node];
  atomicAdd(&sums[g * HCH + j], h[i]);
  if (j == 0) atomicAdd(&cnt[g], 1.0f);
}
__global__ void pool_div_kernel(const float* __restrict__ sums, const float* __restrict__ cnt,
                                float* pooled) {
  int i = blockIdx.x * blockDim.x + threadIdx.x;
  if (i >= GB * HCH) return;
  int g = i >> 7;
  pooled[i] = sums[i] / fmaxf(cnt[g], 1.0f);
}

// ---------------- LSTM (single step, h0=c0=0 so W_hh drops out) + FC ----------------
__global__ void lstm_elem_kernel(const float* __restrict__ gates,
                                 const float* __restrict__ b_ih, const float* __restrict__ b_hh,
                                 float* hn) {
  int i = blockIdx.x * blockDim.x + threadIdx.x;
  if (i >= GB * LHD) return;
  int g = i >> 7, j = i & 127;
  const float* gr = gates + (size_t)g * NGATE;
  float iv = gr[j]       + b_ih[j]       + b_hh[j];
  float gv = gr[256 + j] + b_ih[256 + j] + b_hh[256 + j];
  float ov = gr[384 + j] + b_ih[384 + j] + b_hh[384 + j];
  float si = 1.0f / (1.0f + __expf(-iv));
  float so = 1.0f / (1.0f + __expf(-ov));
  float c = si * tanhf(gv);
  hn[i] = so * tanhf(c);
}
__global__ void fc_kernel(const float* __restrict__ hn, const float* __restrict__ W_fc,
                          const float* __restrict__ b_fc, float* out) {
  int i = blockIdx.x * blockDim.x + threadIdx.x;
  if (i >= GB * OUTD) return;
  int g = i / OUTD, o = i % OUTD;
  const float* hr = hn + (size_t)g * LHD;
  const float* wr = W_fc + (size_t)o * LHD;
  float acc = b_fc[o];
#pragma unroll 8
  for (int k = 0; k < LHD; ++k) acc = fmaf(hr[k], wr[k], acc);
  out[i] = acc;
}

// ---------------- host ----------------
extern "C" void kernel_launch(void* const* d_in, const int* in_sizes, int n_in,
                              void* d_out, int out_size, void* d_ws, size_t ws_size,
                              hipStream_t stream) {
  const float* x      = (const float*)d_in[0];
  const int*   eidx   = (const int*)  d_in[1];
  const int*   batch  = (const int*)  d_in[2];
  const float* Ws     = (const float*)d_in[3];
  const float* bs     = (const float*)d_in[4];
  const float* gammas = (const float*)d_in[5];
  const float* betas  = (const float*)d_in[6];
  const float* rms    = (const float*)d_in[7];
  const float* rvs    = (const float*)d_in[8];
  const float* W_ih   = (const float*)d_in[9];
  /* d_in[10] = W_hh: unused, h0 == 0 */
  const float* b_ih   = (const float*)d_in[11];
  const float* b_hh   = (const float*)d_in[12];
  const float* W_fc   = (const float*)d_in[13];
  const float* b_fc   = (const float*)d_in[14];
  float* out = (float*)d_out;

  const int N = in_sizes[0] / DCH;
  const int E = in_sizes[1] / 2;
  const int* srcI = eidx;
  const int* dstI = eidx + E;

  // carve workspace (256B aligned chunks)
  char* p = (char*)d_ws;
  auto take = [&](size_t bytes) -> char* {
    char* r = (char*)(((uintptr_t)p + 255) & ~(uintptr_t)255);
    p = r + bytes;
    return r;
  };
  size_t nh = (size_t)N * HCH;
  float* B0   = (float*)take(nh * 4);
  float* B1   = (float*)take(nh * 4);
  float* dinv = (float*)take((size_t)N * 4);
  unsigned short* WhiT  = (unsigned short*)take((size_t)NLAYER * DCH * HCH * 2);
  unsigned short* WloT  = (unsigned short*)take((size_t)NLAYER * DCH * HCH * 2);
  unsigned short* WihHi = (unsigned short*)take((size_t)NGATE * HCH * 2);
  unsigned short* WihLo = (unsigned short*)take((size_t)NGATE * HCH * 2);
  float* sums   = (float*)take((size_t)GB * HCH * 4);
  float* cnt    = (float*)take((size_t)GB * 4);
  float* pooled = (float*)take((size_t)GB * HCH * 4);
  float* gates  = (float*)take((size_t)GB * NGATE * 4);
  float* hn     = (float*)take((size_t)GB * LHD * 4);

  const int T = 256;
  auto gs = [](long long n, int t) { return (unsigned int)((n + t - 1) / t); };

  // degree -> dinv (in place)
  init_deg_kernel   <<<gs(N, T), T, 0, stream>>>(dinv, N);
  accum_deg_kernel  <<<gs(E, T), T, 0, stream>>>(dstI, dinv, E);
  finish_dinv_kernel<<<gs(N, T), T, 0, stream>>>(dinv, N);

  // weight pre-splits
  split_transpose_ws_kernel<<<gs((long long)NLAYER * DCH * HCH, T), T, 0, stream>>>(Ws, WhiT, WloT);
  split_plain_kernel<<<gs((long long)NGATE * HCH, T), T, 0, stream>>>(W_ih, WihHi, WihLo, NGATE * HCH);

  // GCN layers (ping-pong B0/B1)
  const float* hin = x;
  float* gOut = B0;
  float* aggB = B1;
  for (int l = 0; l < NLAYER; ++l) {
    // grid: (row tiles) x (groups of 8 col tiles); HCH=128 -> 1 group
    unsigned int gblocks = (unsigned int)((N / 16) * (((HCH / 16) + 7) / 8));
    gemm_wmma_bf16x2_kernel<<<gblocks, 256, 0, stream>>>(
        hin, WhiT + (size_t)l * DCH * HCH, WloT + (size_t)l * DCH * HCH, gOut, N, HCH);
    bias_init_kernel<<<gs((long long)nh, T), T, 0, stream>>>(aggB, bs + l * HCH, (int)nh);
    scatter_edges_kernel<<<gs((long long)(E + N) * 32, T), T, 0, stream>>>(
        srcI, dstI, dinv, gOut, aggB, E, N);
    bn_relu_kernel<<<gs((long long)nh, T), T, 0, stream>>>(
        aggB, gammas + l * HCH, betas + l * HCH, rms + l * HCH, rvs + l * HCH, gOut, (int)nh);
    hin = gOut;
    float* t = gOut; gOut = aggB; aggB = t;
  }

  // pooling
  pool_zero_kernel <<<gs(GB * HCH + GB, T), T, 0, stream>>>(sums, cnt);
  pool_accum_kernel<<<gs((long long)nh, T), T, 0, stream>>>(hin, batch, sums, cnt, (int)nh);
  pool_div_kernel  <<<gs(GB * HCH, T), T, 0, stream>>>(sums, cnt, pooled);

  // LSTM gates GEMM: pooled[128x128] @ W_ih^T[128x512]
  unsigned int ggblocks = (unsigned int)((GB / 16) * (((NGATE / 16) + 7) / 8));
  gemm_wmma_bf16x2_kernel<<<ggblocks, 256, 0, stream>>>(pooled, WihHi, WihLo, gates, GB, NGATE);
  lstm_elem_kernel<<<gs(GB * LHD, T), T, 0, stream>>>(gates, b_ih, b_hh, hn);
  fc_kernel       <<<gs(GB * OUTD, T), T, 0, stream>>>(hn, W_fc, b_fc, out);
}
